// QwenBlock_83391085019788
// MI455X (gfx1250) — compile-verified, matrix operand order checked by reading, not run on hardware
//
#include <hip/hip_runtime.h>
#include <hip/hip_bf16.h>
#include <stdint.h>

// ---------------------------------------------------------------------------
// Qwen block for MI455X (gfx1250, wave32, WMMA).
// All matmuls run on v_wmma_f32_16x16x32_bf16 (bf16 in, f32 accum).
// Weights (fp32) are converted to bf16 while staged into LDS (read once).
// Attention is flash-style (online softmax), causal mask computed inline.
// __launch_bounds__(256) on every kernel: 256-thread blocks = 8 wave32 waves,
// which lets the compiler keep Q fragments / accumulators resident instead of
// spilling to scratch (observed "Folded Reload" spills without it).
// ---------------------------------------------------------------------------

typedef __attribute__((ext_vector_type(16))) __bf16 v16bf;
typedef __attribute__((ext_vector_type(8)))  float  v8f;

#define S_LEN  2048
#define DMODEL 4096
#define NHEAD  32
#define DHEAD  128
#define DFF    11008
#define QKVW   (3 * NHEAD * DHEAD)   // 12288

__device__ __forceinline__ v8f wmma_bf16(v16bf a, v16bf b, v8f c) {
  // D = A(16x32 bf16) * B(32x16 bf16) + C(16x16 f32)
  return __builtin_amdgcn_wmma_f32_16x16x32_bf16(
      /*neg_a=*/false, a, /*neg_b=*/false, b,
      /*c_mod=*/(short)0, c, /*reuse_a=*/false, /*reuse_b=*/false);
}

__device__ __forceinline__ v8f v8f_zero() {
  v8f z;
#pragma unroll
  for (int i = 0; i < 8; ++i) z[i] = 0.0f;
  return z;
}

// ---------------------------------------------------------------------------
// RMSNorm: fp32 in -> bf16 out (GEMM operand).  One block per row.
// ---------------------------------------------------------------------------
__global__ void __launch_bounds__(256)
rmsnorm_kernel(const float* __restrict__ x, const float* __restrict__ w,
               __bf16* __restrict__ out, int cols) {
  const int row = blockIdx.x;
  const int t = threadIdx.x;
  __shared__ float red[256];
  float ss = 0.0f;
  for (int c = t; c < cols; c += 256) {
    float v = x[(size_t)row * cols + c];
    ss += v * v;
  }
  red[t] = ss;
  __syncthreads();
  for (int s2 = 128; s2 > 0; s2 >>= 1) {
    if (t < s2) red[t] += red[t + s2];
    __syncthreads();
  }
  const float inv = rsqrtf(red[0] / (float)cols + 1e-5f);
  for (int c = t; c < cols; c += 256)
    out[(size_t)row * cols + c] = (__bf16)(x[(size_t)row * cols + c] * inv * w[c]);
}

// ---------------------------------------------------------------------------
// Generic WMMA GEMM:  out[M x N] = A[M x K](bf16) * W[N x K](f32)^T (+bias)(+res)
// Block tile 128x128, K-tile 32.  8 waves, each wave owns 64(M) x 32(N)
// = 4x2 WMMA 16x16 accumulators.
// ---------------------------------------------------------------------------
template <bool BIAS, bool RES, bool OUTBF>
__global__ void __launch_bounds__(256)
gemm_kernel(const __bf16* __restrict__ A, const float* __restrict__ W,
            const float* __restrict__ bias, const float* __restrict__ res,
            void* __restrict__ outp, int N, int K) {
  __shared__ __attribute__((aligned(32))) __bf16 As[128][32];
  __shared__ __attribute__((aligned(32))) __bf16 Ws[128][32];

  const int t = threadIdx.x;
  const int w = t >> 5;
  const int lane = t & 31;
  const int lhalf = lane >> 4;   // which K-half of the 32-wide window
  const int l16 = lane & 15;
  const int bn = blockIdx.x * 128;
  const int bm = blockIdx.y * 128;
  const int wm = (w >> 2) * 64;  // wave M offset within block tile
  const int wn = (w & 3) * 32;   // wave N offset within block tile

  v8f acc[4][2];
#pragma unroll
  for (int mi = 0; mi < 4; ++mi)
#pragma unroll
    for (int ni = 0; ni < 2; ++ni) acc[mi][ni] = v8f_zero();

  const int ksteps = K >> 5;
  for (int kt = 0; kt < ksteps; ++kt) {
    // Stage A tile (bf16): 128x32 = 512 uint4, 2 per thread.
#pragma unroll
    for (int i = 0; i < 2; ++i) {
      int idx = t + i * 256;
      int r = idx >> 2;
      int k0 = (idx & 3) * 8;
      *(uint4*)&As[r][k0] =
          *(const uint4*)(A + (size_t)(bm + r) * K + kt * 32 + k0);
    }
    // Stage W tile with fp32 -> bf16 conversion: 128x32 = 1024 float4.
#pragma unroll
    for (int i = 0; i < 4; ++i) {
      int idx = t + i * 256;
      int r = idx >> 3;
      int k0 = (idx & 7) * 4;
      float4 f = *(const float4*)(W + (size_t)(bn + r) * K + kt * 32 + k0);
      Ws[r][k0 + 0] = (__bf16)f.x;
      Ws[r][k0 + 1] = (__bf16)f.y;
      Ws[r][k0 + 2] = (__bf16)f.z;
      Ws[r][k0 + 3] = (__bf16)f.w;
    }
    if (kt + 1 < ksteps) {  // hint the next K-tile into cache
      __builtin_prefetch(A + (size_t)(bm + (t >> 1)) * K + (kt + 1) * 32, 0, 0);
      __builtin_prefetch(W + (size_t)(bn + (t >> 1)) * K + (kt + 1) * 32, 0, 0);
    }
    __syncthreads();

    v16bf Bf[2];
#pragma unroll
    for (int ni = 0; ni < 2; ++ni)
      Bf[ni] = *(const v16bf*)&Ws[wn + ni * 16 + l16][lhalf * 16];
#pragma unroll
    for (int mi = 0; mi < 4; ++mi) {
      v16bf Af = *(const v16bf*)&As[wm + mi * 16 + l16][lhalf * 16];
#pragma unroll
      for (int ni = 0; ni < 2; ++ni)
        acc[mi][ni] = wmma_bf16(Af, Bf[ni], acc[mi][ni]);
    }
    __syncthreads();
  }

  // Epilogue: C layout row = vgpr + 8*(lane/16), col = lane%16.
#pragma unroll
  for (int mi = 0; mi < 4; ++mi)
#pragma unroll
    for (int ni = 0; ni < 2; ++ni)
#pragma unroll
      for (int r = 0; r < 8; ++r) {
        int row = bm + wm + mi * 16 + r + 8 * lhalf;
        int col = bn + wn + ni * 16 + l16;
        float v = acc[mi][ni][r];
        if (BIAS) v += bias[col];
        if (RES) v += res[(size_t)row * N + col];
        if (OUTBF)
          ((__bf16*)outp)[(size_t)row * N + col] = (__bf16)v;
        else
          ((float*)outp)[(size_t)row * N + col] = v;
      }
}

// ---------------------------------------------------------------------------
// RoPE applied in place to bf16 q and k inside the qkv buffer (S x 12288).
// One thread per (s, head, i<64), handles q and k.
// ---------------------------------------------------------------------------
__global__ void __launch_bounds__(256)
rope_kernel(__bf16* __restrict__ qkv, const float* __restrict__ fr,
            const float* __restrict__ fi) {
  int idx = blockIdx.x * blockDim.x + threadIdx.x;  // S*H*64 threads
  int i = idx & 63;
  int h = (idx >> 6) & 31;
  int s = idx >> 11;
  float c = fr[s * 64 + i];
  float sn = fi[s * 64 + i];
  __bf16* q = qkv + (size_t)s * QKVW + h * DHEAD;
  float qr = (float)q[i], qi = (float)q[i + 64];
  q[i] = (__bf16)(qr * c - qi * sn);
  q[i + 64] = (__bf16)(qr * sn + qi * c);
  __bf16* k = q + DMODEL;  // k section offset 4096
  float kr = (float)k[i], ki = (float)k[i + 64];
  k[i] = (__bf16)(kr * c - ki * sn);
  k[i + 64] = (__bf16)(kr * sn + ki * c);
}

// ---------------------------------------------------------------------------
// Flash attention (causal).  Block = (1 head, 128 queries), 8 waves; each
// wave owns 16 query rows.  Key blocks of 64 streamed through LDS; V staged
// transposed so P*V B-fragments are contiguous LDS reads.
// ---------------------------------------------------------------------------
__global__ void __launch_bounds__(256)
flash_attn_kernel(const __bf16* __restrict__ qkv, __bf16* __restrict__ out) {
  const int qt = blockIdx.x;  // query tile (128 rows)
  const int h = blockIdx.y;   // head
  const int t = threadIdx.x;
  const int w = t >> 5;
  const int lane = t & 31;
  const int lhalf = lane >> 4;
  const int l16 = lane & 15;

  __shared__ __attribute__((aligned(32))) __bf16 Kt[64][128];   // [key][dh]
  __shared__ __attribute__((aligned(32))) __bf16 Vt[128][64];   // [dh][key]
  __shared__ __attribute__((aligned(32))) __bf16 Pb[8][16][64]; // per-wave P

  const int qrow0 = qt * 128 + w * 16;

  // Q fragments for this wave (A layout): dh in 4 windows of 32.
  v16bf Qf[4];
#pragma unroll
  for (int ks = 0; ks < 4; ++ks)
    Qf[ks] = *(const v16bf*)(qkv + (size_t)(qrow0 + l16) * QKVW + h * DHEAD +
                             ks * 32 + lhalf * 16);

  v8f O[8];
  float m[8], l[8];
#pragma unroll
  for (int dt = 0; dt < 8; ++dt) O[dt] = v8f_zero();
#pragma unroll
  for (int r = 0; r < 8; ++r) { m[r] = -3.0e38f; l[r] = 0.0f; }

  const float SCALE = 0.08838834764831845f;  // 1/sqrt(128)
  const int nkb = 2 * qt + 2;                // key blocks covering keys <= q_max

  for (int kb = 0; kb < nkb; ++kb) {
    // Stage K (row-major) and V (transposed): 64x128 bf16 each.
#pragma unroll
    for (int i = 0; i < 4; ++i) {
      int idx = t + i * 256;
      int key = idx >> 4;
      int d0 = (idx & 15) * 8;
      const __bf16* base = qkv + (size_t)(kb * 64 + key) * QKVW + h * DHEAD + d0;
      uint4 kv = *(const uint4*)(base + DMODEL);      // K section
      *(uint4*)&Kt[key][d0] = kv;
      uint4 vv = *(const uint4*)(base + 2 * DMODEL);  // V section
      const __bf16* vb = (const __bf16*)&vv;
#pragma unroll
      for (int d = 0; d < 8; ++d) Vt[d0 + d][key] = vb[d];
    }
    __syncthreads();

    // S = Q K^T  (16 q rows x 64 keys per wave)
    v8f S0[4];
#pragma unroll
    for (int kt = 0; kt < 4; ++kt) {
      v8f acc = v8f_zero();
#pragma unroll
      for (int ks = 0; ks < 4; ++ks) {
        v16bf Bf = *(const v16bf*)&Kt[kt * 16 + l16][ks * 32 + lhalf * 16];
        acc = wmma_bf16(Qf[ks], Bf, acc);
      }
      S0[kt] = acc;
    }

    // Scale + causal mask (only blocks touching the diagonal need it).
    const bool diag = (kb >= 2 * qt);
#pragma unroll
    for (int kt = 0; kt < 4; ++kt) {
      int key = kb * 64 + kt * 16 + l16;
#pragma unroll
      for (int r = 0; r < 8; ++r) {
        float v = S0[kt][r] * SCALE;
        if (diag) {
          int qrow = qrow0 + r + 8 * lhalf;
          if (key > qrow) v = -1.0e30f;
        }
        S0[kt][r] = v;
      }
    }

    // Online softmax: row reductions across the 16-lane half (N dimension).
    float mb[8];
#pragma unroll
    for (int r = 0; r < 8; ++r) {
      float v = fmaxf(fmaxf(S0[0][r], S0[1][r]), fmaxf(S0[2][r], S0[3][r]));
      v = fmaxf(v, __shfl_xor(v, 1));
      v = fmaxf(v, __shfl_xor(v, 2));
      v = fmaxf(v, __shfl_xor(v, 4));
      v = fmaxf(v, __shfl_xor(v, 8));
      mb[r] = v;
    }
#pragma unroll
    for (int r = 0; r < 8; ++r) {
      float mnew = fmaxf(m[r], mb[r]);
      float sc = __expf(m[r] - mnew);
      float rs = 0.0f;
#pragma unroll
      for (int kt = 0; kt < 4; ++kt) {
        float p = __expf(S0[kt][r] - mnew);
        S0[kt][r] = p;
        rs += p;
      }
      rs += __shfl_xor(rs, 1);
      rs += __shfl_xor(rs, 2);
      rs += __shfl_xor(rs, 4);
      rs += __shfl_xor(rs, 8);
      l[r] = l[r] * sc + rs;
      m[r] = mnew;
#pragma unroll
      for (int dt = 0; dt < 8; ++dt) O[dt][r] *= sc;
    }

    // P: C layout -> A layout via per-wave LDS scratch.
#pragma unroll
    for (int kt = 0; kt < 4; ++kt)
#pragma unroll
      for (int r = 0; r < 8; ++r)
        Pb[w][r + 8 * lhalf][kt * 16 + l16] = (__bf16)S0[kt][r];
    asm volatile("s_wait_dscnt 0x0" ::: "memory");  // same-wave LDS RAW

    // O += P * V
#pragma unroll
    for (int kt2 = 0; kt2 < 2; ++kt2) {
      v16bf Pf = *(const v16bf*)&Pb[w][l16][kt2 * 32 + lhalf * 16];
#pragma unroll
      for (int dt = 0; dt < 8; ++dt) {
        v16bf Vf = *(const v16bf*)&Vt[dt * 16 + l16][kt2 * 32 + lhalf * 16];
        O[dt] = wmma_bf16(Pf, Vf, O[dt]);
      }
    }
    __syncthreads();  // protect Kt/Vt before next block's staging
  }

  // Normalize and write (S x 4096 bf16).
#pragma unroll
  for (int dt = 0; dt < 8; ++dt)
#pragma unroll
    for (int r = 0; r < 8; ++r) {
      int row = qrow0 + r + 8 * lhalf;
      int col = h * DHEAD + dt * 16 + l16;
      out[(size_t)row * DMODEL + col] = (__bf16)(O[dt][r] / l[r]);
    }
}

// ---------------------------------------------------------------------------
// SwiGLU combine: gu = silu(gate) * up  (bf16 -> bf16, written over gate buf)
// ---------------------------------------------------------------------------
__global__ void __launch_bounds__(256)
silu_mul_kernel(const __bf16* __restrict__ g, const __bf16* __restrict__ u,
                __bf16* __restrict__ out, int n) {
  int i = blockIdx.x * blockDim.x + threadIdx.x;
  if (i < n) {
    float gv = (float)g[i];
    float s = gv / (1.0f + __expf(-gv));
    out[i] = (__bf16)(s * (float)u[i]);
  }
}

// ---------------------------------------------------------------------------
// Launch sequence
// ---------------------------------------------------------------------------
extern "C" void kernel_launch(void* const* d_in, const int* in_sizes, int n_in,
                              void* d_out, int out_size, void* d_ws,
                              size_t ws_size, hipStream_t stream) {
  (void)in_sizes; (void)n_in; (void)out_size; (void)ws_size;

  const float* x     = (const float*)d_in[0];
  const float* fr    = (const float*)d_in[1];
  const float* fi    = (const float*)d_in[2];
  /* d_in[3] attention_mask: causal mask computed inline */
  const float* ln1   = (const float*)d_in[4];
  const float* qkv_w = (const float*)d_in[5];
  const float* qkv_b = (const float*)d_in[6];
  const float* o_w   = (const float*)d_in[7];
  const float* ln2   = (const float*)d_in[8];
  const float* wg    = (const float*)d_in[9];
  const float* wi    = (const float*)d_in[10];
  const float* wo    = (const float*)d_in[11];

  // Workspace layout (~208 MB), all offsets 256B-aligned.
  char* ws = (char*)d_ws;
  __bf16* h_bf = (__bf16*)ws;  ws += (size_t)S_LEN * DMODEL * 2;  // 16 MB
  __bf16* qkv  = (__bf16*)ws;  ws += (size_t)S_LEN * QKVW * 2;    // 48 MB
  __bf16* attn = (__bf16*)ws;  ws += (size_t)S_LEN * DMODEL * 2;  // 16 MB
  float*  x1   = (float*)ws;   ws += (size_t)S_LEN * DMODEL * 4;  // 32 MB
  __bf16* gate = (__bf16*)ws;  ws += (size_t)S_LEN * DFF * 2;     // 45 MB
  __bf16* up   = (__bf16*)ws;                                     // 45 MB

  const dim3 blk(256);

  // --- attention ---
  rmsnorm_kernel<<<dim3(S_LEN), blk, 0, stream>>>(x, ln1, h_bf, DMODEL);

  gemm_kernel<true, false, true><<<dim3(QKVW / 128, S_LEN / 128), blk, 0, stream>>>(
      h_bf, qkv_w, qkv_b, nullptr, qkv, QKVW, DMODEL);

  rope_kernel<<<dim3((S_LEN * NHEAD * 64) / 256), blk, 0, stream>>>(qkv, fr, fi);

  flash_attn_kernel<<<dim3(S_LEN / 128, NHEAD), blk, 0, stream>>>(qkv, attn);

  gemm_kernel<false, true, false><<<dim3(DMODEL / 128, S_LEN / 128), blk, 0, stream>>>(
      attn, o_w, nullptr, x, x1, DMODEL, DMODEL);

  // --- SwiGLU FFN ---
  rmsnorm_kernel<<<dim3(S_LEN), blk, 0, stream>>>(x1, ln2, h_bf, DMODEL);

  gemm_kernel<false, false, true><<<dim3(DFF / 128, S_LEN / 128), blk, 0, stream>>>(
      h_bf, wg, nullptr, nullptr, gate, DFF, DMODEL);

  gemm_kernel<false, false, true><<<dim3(DFF / 128, S_LEN / 128), blk, 0, stream>>>(
      h_bf, wi, nullptr, nullptr, up, DFF, DMODEL);

  silu_mul_kernel<<<dim3((S_LEN * DFF) / 256), blk, 0, stream>>>(
      gate, up, gate, S_LEN * DFF);

  gemm_kernel<false, true, false><<<dim3(DMODEL / 128, S_LEN / 128), blk, 0, stream>>>(
      gate, wo, nullptr, x1, (float*)d_out, DMODEL, DFF);
}
